// StructureModule_49435073576971
// MI455X (gfx1250) — matrix operand order repaired and checked by reading, not compiled, not run on hardware
//
#include <hip/hip_runtime.h>

typedef __attribute__((ext_vector_type(16))) _Float16 v16h;
typedef __attribute__((ext_vector_type(8)))  float    v8f;
typedef __attribute__((ext_vector_type(4)))  float    v4f;
typedef __attribute__((ext_vector_type(4)))  _Float16 h4;

#define NRES  512
#define DS    384      // CS
#define DZ    128      // CZ
#define NH    12       // heads
#define CH    16       // C per head
#define NPQ   4
#define NPV   8
#define NBLK  8
#define KAUG  32       // fused attention inner dim: 16 qk + 12 pts + 2 norms + 2 pad
#define VCW   64       // combined V row width (16 v + 24 v_pts + 24 zero pad -> interior tile)
#define CATW  2112     // 192 + 288 + 96 + 1536
#define PROJW 1152     // 192 + 384 + 144 + 432

// ---- WMMA f16 fragment-layout maps (ISA 7.12.2), used for LDS/global staging ----
// A (16x32, MxK): lane = m + ((k>>3)&1)*16 ; elem = (k>>4)*8 + (k&7)
// B (32x16, KxN): lane = n + ((k>>3)&1)*16 ; elem = (k>>4)*8 + (k&7)
__device__ __forceinline__ int aLane(int m, int k) { return m + (((k >> 3) & 1) << 4); }
__device__ __forceinline__ int aElem(int k)        { return ((k >> 4) << 3) + (k & 7); }
__device__ __forceinline__ int bLane(int n, int k) { return n + (((k >> 3) & 1) << 4); }
__device__ __forceinline__ int bElem(int k)        { return ((k >> 4) << 3) + (k & 7); }

// ---------------- wave32 reductions ----------------
__device__ __forceinline__ float wsum(float v) {
#pragma unroll
  for (int o = 16; o > 0; o >>= 1) v += __shfl_xor(v, o, 32);
  return v;
}
__device__ __forceinline__ float wmaxr(float v) {
#pragma unroll
  for (int o = 16; o > 0; o >>= 1) v = fmaxf(v, __shfl_xor(v, o, 32));
  return v;
}

// ---------------- generic batched GEMM: C = A@B (+bias)(+res)(relu), f16 WMMA, f32 acc --
// 256 threads = 8 waves; 64x64 tile; LDS holds tiles in fragment order so each wave
// fetches v16h fragments with ds_load_b128 pairs. Requires K % 32 == 0.
__global__ __launch_bounds__(256) void k_gemm(
    const float* __restrict__ A, int lda, long long sA,
    const float* __restrict__ B, int ldb, long long sB,
    const float* __restrict__ bias,
    const float* __restrict__ res, int ldr, long long sR,
    float* __restrict__ C, int ldc, long long sC,
    int M, int N, int K, int relu)
{
  __shared__ __align__(16) _Float16 Af[4][32][16];   // [m-tile][lane][elem]
  __shared__ __align__(16) _Float16 Bf[4][32][16];   // [n-tile][lane][elem]
  const int bz = blockIdx.z;
  A += sA * bz; B += sB * bz; C += sC * bz;
  if (res) res += sR * bz;
  const int m0 = blockIdx.x * 64, n0 = blockIdx.y * 64;
  const int t = threadIdx.x, lane = t & 31, wv = t >> 5;
  const int wr  = wv & 3;            // wave tile-row 0..3
  const int wc0 = (wv >> 2) << 1;    // wave tile-col base 0 or 2
  const bool inM = (m0 + 64 <= M);
  const bool inN = (n0 + 64 <= N);
  v8f acc0 = {}, acc1 = {};
  for (int k0 = 0; k0 < K; k0 += 32) {
    if (inM) {                                   // vectorized A panel 64x32
#pragma unroll
      for (int j = 0; j < 2; ++j) {
        int li = t + j * 256;
        int r = li >> 3, k4 = (li & 7) << 2;     // 4 consecutive k -> same lane, contig elems
        const float* ap = A + (long long)(m0 + r) * lda + (k0 + k4);
        v4f v = *(const v4f*)ap;
        if (k0 + 32 < K) __builtin_prefetch(ap + 32, 0, 3);   // next K-chunk
        h4 hv = __builtin_convertvector(v, h4);
        *(h4*)&Af[r >> 4][aLane(r & 15, k4)][aElem(k4)] = hv;
      }
    } else {
#pragma unroll
      for (int j = 0; j < 8; ++j) {
        int li = t + j * 256;
        int r = li >> 5, kk = li & 31;
        int gm = m0 + r;
        float v = (gm < M) ? A[(long long)gm * lda + (k0 + kk)] : 0.f;
        Af[r >> 4][aLane(r & 15, kk)][aElem(kk)] = (_Float16)v;
      }
    }
    if (inN) {                                   // B panel 32x64, 4 k-rows per thread
#pragma unroll
      for (int j = 0; j < 2; ++j) {
        int li = t + j * 256;
        int n = li & 63, k4 = (li >> 6) << 2;
        const float* bp = B + (long long)(k0 + k4) * ldb + (n0 + n);
        v4f v;
        v[0] = bp[0]; v[1] = bp[ldb]; v[2] = bp[2 * ldb]; v[3] = bp[3 * ldb];
        if (k0 + 32 < K) __builtin_prefetch(bp + (long long)32 * ldb, 0, 3);
        h4 hv = __builtin_convertvector(v, h4);
        *(h4*)&Bf[n >> 4][bLane(n & 15, k4)][bElem(k4)] = hv;
      }
    } else {
#pragma unroll
      for (int j = 0; j < 8; ++j) {
        int li = t + j * 256;
        int kk = li >> 6, n = li & 63;
        int gn = n0 + n;
        float v = (gn < N) ? B[(long long)(k0 + kk) * ldb + gn] : 0.f;
        Bf[n >> 4][bLane(n & 15, kk)][bElem(kk)] = (_Float16)v;
      }
    }
    __syncthreads();
    v16h af = *(const v16h*)&Af[wr][lane][0];
    v16h b0 = *(const v16h*)&Bf[wc0][lane][0];
    v16h b1 = *(const v16h*)&Bf[wc0 + 1][lane][0];
    acc0 = __builtin_amdgcn_wmma_f32_16x16x32_f16(false, af, false, b0, (short)0, acc0, false, false);
    acc1 = __builtin_amdgcn_wmma_f32_16x16x32_f16(false, af, false, b1, (short)0, acc1, false, false);
    __syncthreads();
  }
  const int ml = lane & 15;
  const int rb = (lane < 16) ? 0 : 8;
#pragma unroll
  for (int r = 0; r < 8; ++r) {
    int gm = m0 + wr * 16 + rb + r;
    if (gm >= M) continue;
#pragma unroll
    for (int hh = 0; hh < 2; ++hh) {
      int gn = n0 + (wc0 + hh) * 16 + ml;
      if (gn >= N) continue;
      float v = hh ? acc1[r] : acc0[r];
      if (bias) v += bias[gn];
      if (res)  v += res[(long long)gm * ldr + gn];
      if (relu) v = fmaxf(v, 0.f);
      C[(long long)gm * ldc + gn] = v;
    }
  }
}

// ---------------- LayerNorm (wave per row) ----------------
__global__ __launch_bounds__(256) void k_layernorm(
    const float* __restrict__ x, const float* __restrict__ g,
    const float* __restrict__ b, float* __restrict__ y, int rows, int cols)
{
  int row = blockIdx.x * 8 + (threadIdx.x >> 5);
  int lane = threadIdx.x & 31;
  if (row >= rows) return;
  const float* xr = x + (long long)row * cols;
  float s = 0.f, s2 = 0.f;
  for (int c = lane; c < cols; c += 32) { float v = xr[c]; s += v; s2 += v * v; }
  s = wsum(s); s2 = wsum(s2);
  float m = s / cols;
  float rs = rsqrtf(s2 / cols - m * m + 1e-5f);
  float* yr = y + (long long)row * cols;
  for (int c = lane; c < cols; c += 32) yr[c] = (xr[c] - m) * rs * g[c] + b[c];
}

// ---------------- pair LN stats: mu / rstd per (i,j) row of 128 ----------------
__global__ __launch_bounds__(256) void k_pair_stats(
    const float* __restrict__ pair, float* __restrict__ mu, float* __restrict__ rstd)
{
  long long row = (long long)blockIdx.x * 8 + (threadIdx.x >> 5);
  int lane = threadIdx.x & 31;
  const float* p = pair + row * DZ;
  float s = 0.f, s2 = 0.f;
  for (int c = lane; c < DZ; c += 32) { float v = p[c]; s += v; s2 += v * v; }
  s = wsum(s); s2 = wsum(s2);
  if (lane == 0) {
    float m = s * (1.f / DZ);
    mu[row] = m;
    rstd[row] = rsqrtf(s2 * (1.f / DZ) - m * m + 1e-5f);
  }
}

// ---------------- block-invariant bias2[h][i][j] = sqrt(1/3)*(z.wb+bb) + mbias ----------
__global__ __launch_bounds__(256) void k_bias2(
    const float* __restrict__ pair, const float* __restrict__ mu,
    const float* __restrict__ rstd, const float* __restrict__ zg,
    const float* __restrict__ zb, const float* __restrict__ wb,
    const float* __restrict__ bbv, const float* __restrict__ mask,
    float* __restrict__ bias2)
{
  __shared__ float wbs[DZ * NH];
  __shared__ float zgs[DZ], zbs[DZ];
  for (int c = threadIdx.x; c < DZ * NH; c += 256) wbs[c] = wb[c];
  for (int c = threadIdx.x; c < DZ; c += 256) { zgs[c] = zg[c]; zbs[c] = zb[c]; }
  __syncthreads();
  int i = blockIdx.x;
  int lane = threadIdx.x & 31, wv = threadIdx.x >> 5;
  float mi = mask[i];
  for (int j = wv; j < NRES; j += 8) {
    long long rz = (long long)i * NRES + j;
    const float* p = pair + rz * DZ;
    float m = mu[rz], rs = rstd[rz];
    float acc[NH];
#pragma unroll
    for (int h = 0; h < NH; ++h) acc[h] = 0.f;
    for (int c = lane; c < DZ; c += 32) {
      float z = (p[c] - m) * rs * zgs[c] + zbs[c];
#pragma unroll
      for (int h = 0; h < NH; ++h) acc[h] += z * wbs[c * NH + h];
    }
#pragma unroll
    for (int h = 0; h < NH; ++h) acc[h] = wsum(acc[h]);
    if (lane == 0) {
      float mb = 1e8f * (mi * mask[j] - 1.f);
#pragma unroll
      for (int h = 0; h < NH; ++h)
        bias2[((long long)h * NRES + i) * NRES + j] =
            0.5773502691896258f * (acc[h] + bbv[h]) + mb;
    }
  }
}

// ---------------- pack projection weights into one 384x1152 matrix ----------------
__global__ void k_pack_w(const float* __restrict__ wq, const float* __restrict__ wkv,
                         const float* __restrict__ wqp, const float* __restrict__ wkvp,
                         const float* __restrict__ bq, const float* __restrict__ bkv,
                         const float* __restrict__ bqp, const float* __restrict__ bkvp,
                         float* __restrict__ Wcat, float* __restrict__ Bcat)
{
  int idx = blockIdx.x * blockDim.x + threadIdx.x;
  if (idx >= DS * PROJW) return;
  int r = idx / PROJW, c = idx % PROJW;
  float v;
  if (c < 192)      v = wq[r * 192 + c];
  else if (c < 576) v = wkv[r * 384 + (c - 192)];
  else if (c < 720) v = wqp[r * 144 + (c - 576)];
  else              v = wkvp[r * 432 + (c - 720)];
  Wcat[idx] = v;
  if (idx < PROJW) {
    float bv;
    if (idx < 192)      bv = bq[idx];
    else if (idx < 576) bv = bkv[idx - 192];
    else if (idx < 720) bv = bqp[idx - 576];
    else                bv = bkvp[idx - 720];
    Bcat[idx] = bv;
  }
}

// ------ build augmented QA/KA in f16 *fragment layout* (K=32 fused logits) + Vc --------
__global__ void k_rot_pts(const float* __restrict__ proj,
                          const float* __restrict__ rot, const float* __restrict__ trans,
                          const float* __restrict__ head_w,
                          _Float16* __restrict__ QAh, _Float16* __restrict__ KAh,
                          float* __restrict__ Vc)
{
  int idx = blockIdx.x * blockDim.x + threadIdx.x;
  if (idx >= NRES * NH) return;
  int i = idx / NH, h = idx % NH;
  const float* R = rot + i * 9;
  const float* T = trans + i * 3;
  const float* row = proj + (long long)i * PROJW;
  float hw = logf(1.f + expf(head_w[h])) * 0.13608276348795434f;  // softplus * sqrt(1/54)
  const float s1 = 0.14433756729740643f;                          // sqrt(1/(3*C))
  float qv[KAUG], kv[KAUG];
  float* vc = Vc + ((long long)h * NRES + i) * VCW;
#pragma unroll
  for (int c = 0; c < CH; ++c) {
    qv[c] = row[h * CH + c] * s1;               // q scaled
    kv[c] = row[192 + h * 2 * CH + c];          // k
    vc[c] = row[192 + h * 2 * CH + CH + c];     // v
  }
  float q2 = 0.f;
#pragma unroll
  for (int p = 0; p < NPQ; ++p) {
    float x = row[576 + (h * NPQ + p) * 3 + 0];
    float y = row[576 + (h * NPQ + p) * 3 + 1];
    float z = row[576 + (h * NPQ + p) * 3 + 2];
    float rx = R[0]*x + R[1]*y + R[2]*z + T[0];
    float ry = R[3]*x + R[4]*y + R[5]*z + T[1];
    float rz = R[6]*x + R[7]*y + R[8]*z + T[2];
    qv[16 + p*3 + 0] = hw * rx; qv[16 + p*3 + 1] = hw * ry; qv[16 + p*3 + 2] = hw * rz;
    q2 += rx*rx + ry*ry + rz*rz;
  }
  qv[28] = -0.5f * hw * q2; qv[29] = 1.f; qv[30] = 0.f; qv[31] = 0.f;
  float k2 = 0.f;
#pragma unroll
  for (int p = 0; p < NPQ + NPV; ++p) {
    float x = row[720 + (h * (NPQ + NPV) + p) * 3 + 0];
    float y = row[720 + (h * (NPQ + NPV) + p) * 3 + 1];
    float z = row[720 + (h * (NPQ + NPV) + p) * 3 + 2];
    float rx = R[0]*x + R[1]*y + R[2]*z + T[0];
    float ry = R[3]*x + R[4]*y + R[5]*z + T[1];
    float rz = R[6]*x + R[7]*y + R[8]*z + T[2];
    if (p < NPQ) {
      kv[16 + p*3 + 0] = rx; kv[16 + p*3 + 1] = ry; kv[16 + p*3 + 2] = rz;
      k2 += rx*rx + ry*ry + rz*rz;
    } else {
      int q = p - NPQ;
      vc[16 + q*3 + 0] = rx; vc[16 + q*3 + 1] = ry; vc[16 + q*3 + 2] = rz;
    }
  }
  kv[28] = 1.f; kv[29] = -0.5f * hw * k2; kv[30] = 0.f; kv[31] = 0.f;
#pragma unroll
  for (int c = 40; c < VCW; ++c) vc[c] = 0.f;   // zero pad -> apply-GEMM fully interior
  // scatter into fragment layout: [h][res-tile][lane][elem]
  const long long base = ((long long)h * 32 + (i >> 4)) * 32 * 16;
  const int m = i & 15;
#pragma unroll
  for (int k = 0; k < KAUG; ++k) {
    QAh[base + aLane(m, k) * 16 + aElem(k)] = (_Float16)qv[k];
    KAh[base + bLane(m, k) * 16 + bElem(k)] = (_Float16)kv[k];
  }
}

// ---------------- attention logits: one WMMA per 16x16 tile, fragments direct ----------
__global__ __launch_bounds__(32) void k_attn_logits(
    const _Float16* __restrict__ QAh, const _Float16* __restrict__ KAh,
    const float* __restrict__ bias2, float* __restrict__ logits)
{
  int h = blockIdx.z;
  int i0 = blockIdx.x * 16, j0 = blockIdx.y * 16;
  int lane = threadIdx.x;
  v16h af = *(const v16h*)(QAh + (((long long)h * 32 + blockIdx.x) * 32 + lane) * 16);
  v16h bf = *(const v16h*)(KAh + (((long long)h * 32 + blockIdx.y) * 32 + lane) * 16);
  v8f acc = {};
  acc = __builtin_amdgcn_wmma_f32_16x16x32_f16(false, af, false, bf, (short)0, acc, false, false);
  const int ml = lane & 15;
  const int rb = (lane < 16) ? 0 : 8;
  const float* bh = bias2 + ((long long)h * NRES + i0) * NRES + j0;
  float* outp    = logits + ((long long)h * NRES + i0) * NRES + j0;
#pragma unroll
  for (int r = 0; r < 8; ++r) {
    int rr = rb + r;
    outp[rr * NRES + ml] = acc[r] + bh[rr * NRES + ml];
  }
}

// ---------------- softmax over last dim (wave per row) ----------------
__global__ __launch_bounds__(256) void k_softmax(float* __restrict__ a, int rows, int cols)
{
  long long row = (long long)blockIdx.x * 8 + (threadIdx.x >> 5);
  int lane = threadIdx.x & 31;
  if (row >= rows) return;
  float* p = a + row * cols;
  float mx = -1e30f;
  for (int c = lane; c < cols; c += 32) mx = fmaxf(mx, p[c]);
  mx = wmaxr(mx);
  float s = 0.f;
  for (int c = lane; c < cols; c += 32) { float e = __expf(p[c] - mx); p[c] = e; s += e; }
  s = wsum(s);
  float inv = 1.f / s;
  for (int c = lane; c < cols; c += 32) p[c] *= inv;
}

// ---------------- o_pair: per-residue (12x512)@(512x128), LN(z) on the fly -------------
__global__ __launch_bounds__(256) void k_opair(
    const float* __restrict__ attn, const float* __restrict__ pair,
    const float* __restrict__ mu, const float* __restrict__ rstd,
    const float* __restrict__ zg, const float* __restrict__ zb,
    float* __restrict__ cat)
{
  __shared__ __align__(16) _Float16 Af[32][16];       // one 16x32 A fragment tile
  __shared__ __align__(16) _Float16 Bf[8][32][16];    // 8 n-tiles of 32x16
  int i = blockIdx.x;
  int t = threadIdx.x, lane = t & 31, wv = t >> 5;    // wave = n-tile (8 x 16 cols = 128)
  v8f acc = {};
  for (int k0 = 0; k0 < NRES; k0 += 32) {
#pragma unroll
    for (int j = 0; j < 2; ++j) {
      int li = t + j * 256;
      int r = li >> 5, kk = li & 31;                  // r = head row (0..15)
      float v = (r < NH) ? attn[((long long)r * NRES + i) * NRES + k0 + kk] : 0.f;
      Af[aLane(r, kk)][aElem(kk)] = (_Float16)v;
    }
#pragma unroll
    for (int j = 0; j < 4; ++j) {
      int li = t + j * 256;
      int c = li & 127, kg = li >> 7;                 // 4 k-rows per thread, coalesced in c
      int k4 = kg << 2;
      h4 hv;
#pragma unroll
      for (int d = 0; d < 4; ++d) {
        long long rz = (long long)i * NRES + k0 + k4 + d;
        hv[d] = (_Float16)((pair[rz * DZ + c] - mu[rz]) * rstd[rz] * zg[c] + zb[c]);
      }
      if (k0 + 32 < NRES)                             // prefetch next z chunk out of L2
        __builtin_prefetch(pair + ((long long)i * NRES + k0 + 32 + k4) * DZ + c, 0, 3);
      *(h4*)&Bf[c >> 4][bLane(c & 15, k4)][bElem(k4)] = hv;
    }
    __syncthreads();
    v16h af = *(const v16h*)&Af[lane][0];
    v16h bf = *(const v16h*)&Bf[wv][lane][0];
    acc = __builtin_amdgcn_wmma_f32_16x16x32_f16(false, af, false, bf, (short)0, acc, false, false);
    __syncthreads();
  }
  const int ml = lane & 15;
  const int rb = (lane < 16) ? 0 : 8;
#pragma unroll
  for (int r = 0; r < 8; ++r) {
    int hrow = rb + r;
    if (hrow < NH)
      cat[(long long)i * CATW + 576 + hrow * DZ + wv * 16 + ml] = acc[r];
  }
}

// ---------------- finalize IPA: inverse-rotate o_pt, norms, pack cat -------------------
__global__ void k_finalize(const float* __restrict__ Oc,
                           const float* __restrict__ rot, const float* __restrict__ trans,
                           float* __restrict__ cat)
{
  int idx = blockIdx.x * blockDim.x + threadIdx.x;
  if (idx >= NRES * NH) return;
  int i = idx / NH, h = idx % NH;
  const float* R = rot + i * 9;
  const float* T = trans + i * 3;
  const float* oh = Oc + ((long long)h * NRES + i) * VCW;
  float* cr = cat + (long long)i * CATW;
#pragma unroll
  for (int c = 0; c < CH; ++c) cr[h * CH + c] = oh[c];
#pragma unroll
  for (int p = 0; p < NPV; ++p) {
    float x = oh[16 + p*3 + 0] - T[0];
    float y = oh[16 + p*3 + 1] - T[1];
    float z = oh[16 + p*3 + 2] - T[2];
    float lx = R[0]*x + R[3]*y + R[6]*z;   // R^T
    float ly = R[1]*x + R[4]*y + R[7]*z;
    float lz = R[2]*x + R[5]*y + R[8]*z;
    cr[192 + 0*96 + h*NPV + p] = lx;
    cr[192 + 1*96 + h*NPV + p] = ly;
    cr[192 + 2*96 + h*NPV + p] = lz;
    cr[480 + h*NPV + p] = sqrtf(lx*lx + ly*ly + lz*lz + 1e-6f);
  }
}

// ---------------- backbone: quat update, compose frames, positions ---------------------
__global__ void k_frames(const float* __restrict__ upd,
                         float* __restrict__ rot, float* __restrict__ trans,
                         float* __restrict__ out_r, float* __restrict__ out_t,
                         float* __restrict__ out_p)
{
  int i = blockIdx.x * blockDim.x + threadIdx.x;
  if (i >= NRES) return;
  const float* u = upd + i * 6;
  float b = u[0], c = u[1], d = u[2];
  float n = rsqrtf(1.f + b*b + c*c + d*d);
  float w = n; b *= n; c *= n; d *= n;
  float Ru[9] = {
    1.f - 2.f*(c*c + d*d), 2.f*(b*c - w*d),       2.f*(b*d + w*c),
    2.f*(b*c + w*d),       1.f - 2.f*(b*b + d*d), 2.f*(c*d - w*b),
    2.f*(b*d - w*c),       2.f*(c*d + w*b),       1.f - 2.f*(b*b + c*c)};
  float* R = rot + i * 9;
  float* T = trans + i * 3;
  float tu0 = u[3], tu1 = u[4], tu2 = u[5];
  float Tn[3], Rn[9];
#pragma unroll
  for (int r = 0; r < 3; ++r) {
    Tn[r] = R[r*3+0]*tu0 + R[r*3+1]*tu1 + R[r*3+2]*tu2 + T[r];
#pragma unroll
    for (int cc = 0; cc < 3; ++cc)
      Rn[r*3+cc] = R[r*3+0]*Ru[0*3+cc] + R[r*3+1]*Ru[1*3+cc] + R[r*3+2]*Ru[2*3+cc];
  }
#pragma unroll
  for (int k = 0; k < 9; ++k) { R[k] = Rn[k]; out_r[i*9 + k] = Rn[k]; }
#pragma unroll
  for (int k = 0; k < 3; ++k) { T[k] = Tn[k]; out_t[i*3 + k] = Tn[k]; }
  const float LIT[4][3] = {{-0.525f, 1.363f, 0.f}, {0.f, 0.f, 0.f},
                           {1.526f, 0.f, 0.f},     {0.627f, 1.062f, 0.f}};
#pragma unroll
  for (int a = 0; a < 4; ++a)
#pragma unroll
    for (int x = 0; x < 3; ++x)
      out_p[i*12 + a*3 + x] =
          Rn[x*3+0]*LIT[a][0] + Rn[x*3+1]*LIT[a][1] + Rn[x*3+2]*LIT[a][2] + Tn[x];
}

__global__ void k_init_frames(float* __restrict__ rot, float* __restrict__ trans)
{
  int i = blockIdx.x * blockDim.x + threadIdx.x;
  if (i >= NRES) return;
  float* R = rot + i * 9;
  R[0]=1.f; R[1]=0.f; R[2]=0.f; R[3]=0.f; R[4]=1.f; R[5]=0.f; R[6]=0.f; R[7]=0.f; R[8]=1.f;
  trans[i*3+0]=0.f; trans[i*3+1]=0.f; trans[i*3+2]=0.f;
}

__global__ void k_copy(const float* __restrict__ src, float* __restrict__ dst, int n)
{
  int i = blockIdx.x * blockDim.x + threadIdx.x;
  if (i < n) dst[i] = src[i];
}

// =====================================================================================
enum {
  IN_SINGLE = 0, IN_PAIR, IN_MASK,
  P_LN_S_G, P_LN_S_B, P_LN_Z_G, P_LN_Z_B, P_IN_W, P_IN_B,
  P_WQ, P_BQ, P_WKV, P_BKV, P_WQP, P_BQP, P_WKVP, P_BKVP,
  P_WB, P_BB, P_HEAD_W, P_WO, P_BO,
  P_LN_IPA_G, P_LN_IPA_B, P_LN_T_G, P_LN_T_B, P_BB_W, P_BB_B,
  P_T0_W1, P_T0_B1, P_T0_W2, P_T0_B2, P_T0_W3, P_T0_B3
};

extern "C" void kernel_launch(void* const* d_in, const int* in_sizes, int n_in,
                              void* d_out, int out_size, void* d_ws, size_t ws_size,
                              hipStream_t stream) {
  (void)in_sizes; (void)n_in; (void)out_size; (void)ws_size;
  #define PF(k) ((const float*)d_in[(k)])
  const float* single = PF(IN_SINGLE);
  const float* pair   = PF(IN_PAIR);
  const float* mask   = PF(IN_MASK);

  // ---- workspace layout (floats, offsets 64-aligned), ~43 MB total ----
  float* ws = (float*)d_ws;
  size_t off = 0;
  #define CARVE(nm, cnt) float* nm = ws + off; off = (off + (size_t)(cnt) + 63) & ~(size_t)63
  CARVE(tmp1, (size_t)NRES * DS);
  CARVE(tmp2, (size_t)NRES * DS);
  CARVE(s,    (size_t)NRES * DS);
  CARVE(proj, (size_t)NRES * PROJW);
  CARVE(Wcat, (size_t)DS * PROJW);
  CARVE(Bcat, PROJW);
  CARVE(QAf,  (size_t)NH * NRES * KAUG / 2);   // f16 fragment-layout
  CARVE(KAf,  (size_t)NH * NRES * KAUG / 2);   // f16 fragment-layout
  CARVE(Vc,   (size_t)NH * NRES * VCW);
  CARVE(attn, (size_t)NH * NRES * NRES);
  CARVE(bias2,(size_t)NH * NRES * NRES);
  CARVE(Oc,   (size_t)NH * NRES * VCW);
  CARVE(cat,  (size_t)NRES * CATW);
  CARVE(mu,   (size_t)NRES * NRES);
  CARVE(rstd, (size_t)NRES * NRES);
  CARVE(rot,  (size_t)NRES * 9);
  CARVE(trans,(size_t)NRES * 3);
  CARVE(upd,  (size_t)NRES * 6);
  #undef CARVE
  _Float16* QAh = (_Float16*)QAf;
  _Float16* KAh = (_Float16*)KAf;

  float* out   = (float*)d_out;
  float* out_s = out;
  float* out_r = out + 196608;
  float* out_t = out + 196608 + 36864;
  float* out_p = out + 196608 + 36864 + 12288;

  // ---- prologue ----
  k_pack_w<<<(DS * PROJW + 255) / 256, 256, 0, stream>>>(
      PF(P_WQ), PF(P_WKV), PF(P_WQP), PF(P_WKVP),
      PF(P_BQ), PF(P_BKV), PF(P_BQP), PF(P_BKVP), Wcat, Bcat);
  k_layernorm<<<64, 256, 0, stream>>>(single, PF(P_LN_S_G), PF(P_LN_S_B), tmp1, NRES, DS);
  k_pair_stats<<<NRES * NRES / 8, 256, 0, stream>>>(pair, mu, rstd);
  k_bias2<<<NRES, 256, 0, stream>>>(pair, mu, rstd, PF(P_LN_Z_G), PF(P_LN_Z_B),
                                    PF(P_WB), PF(P_BB), mask, bias2);
  k_gemm<<<dim3(8, 6, 1), 256, 0, stream>>>(tmp1, DS, 0, PF(P_IN_W), DS, 0, PF(P_IN_B),
                                            nullptr, 0, 0, s, DS, 0, NRES, DS, DS, 0);
  k_init_frames<<<2, 256, 0, stream>>>(rot, trans);

  for (int blk = 0; blk < NBLK; ++blk) {
    // fused IPA projections: one fully-interior N=1152 GEMM
    k_gemm<<<dim3(8, 18, 1), 256, 0, stream>>>(s, DS, 0, Wcat, PROJW, 0, Bcat,
        nullptr, 0, 0, proj, PROJW, 0, NRES, PROJW, DS, 0);
    // rotate points, build fragment-layout QA/KA and combined V
    k_rot_pts<<<24, 256, 0, stream>>>(proj, rot, trans, PF(P_HEAD_W), QAh, KAh, Vc);
    // fused logits (qk + point attention + bias) and softmax
    k_attn_logits<<<dim3(32, 32, NH), 32, 0, stream>>>(QAh, KAh, bias2, attn);
    k_softmax<<<NH * NRES / 8, 256, 0, stream>>>(attn, NH * NRES, NRES);
    // o and o_pt: batched per-head GEMM a[h] @ Vc[h] (Vc zero-padded to 64 -> interior)
    k_gemm<<<dim3(8, 1, NH), 256, 0, stream>>>(
        attn, NRES, (long long)NRES * NRES, Vc, VCW, (long long)NRES * VCW,
        nullptr, nullptr, 0, 0, Oc, VCW, (long long)NRES * VCW, NRES, VCW, NRES, 0);
    // o_pair straight into cat
    k_opair<<<NRES, 256, 0, stream>>>(attn, pair, mu, rstd, PF(P_LN_Z_G), PF(P_LN_Z_B), cat);
    k_finalize<<<24, 256, 0, stream>>>(Oc, rot, trans, cat);
    // s = LN(s + cat @ wo + bo)
    k_gemm<<<dim3(8, 6, 1), 256, 0, stream>>>(cat, CATW, 0, PF(P_WO), DS, 0, PF(P_BO),
        s, DS, 0, s, DS, 0, NRES, DS, CATW, 0);
    k_layernorm<<<64, 256, 0, stream>>>(s, PF(P_LN_IPA_G), PF(P_LN_IPA_B), s, NRES, DS);
    // transition (NTL = 1)
    k_gemm<<<dim3(8, 6, 1), 256, 0, stream>>>(s, DS, 0, PF(P_T0_W1), DS, 0, PF(P_T0_B1),
        nullptr, 0, 0, tmp1, DS, 0, NRES, DS, DS, 1);
    k_gemm<<<dim3(8, 6, 1), 256, 0, stream>>>(tmp1, DS, 0, PF(P_T0_W2), DS, 0, PF(P_T0_B2),
        nullptr, 0, 0, tmp2, DS, 0, NRES, DS, DS, 1);
    k_gemm<<<dim3(8, 6, 1), 256, 0, stream>>>(tmp2, DS, 0, PF(P_T0_W3), DS, 0, PF(P_T0_B3),
        s, DS, 0, s, DS, 0, NRES, DS, DS, 0);
    k_layernorm<<<64, 256, 0, stream>>>(s, PF(P_LN_T_G), PF(P_LN_T_B), s, NRES, DS);
    // backbone update + frames + positions
    k_gemm<<<dim3(8, 1, 1), 256, 0, stream>>>(s, DS, 0, PF(P_BB_W), 6, 0, PF(P_BB_B),
        nullptr, 0, 0, upd, 6, 0, NRES, 6, DS, 0);
    k_frames<<<2, 256, 0, stream>>>(upd, rot, trans,
                                    out_r + (size_t)blk * NRES * 9,
                                    out_t + (size_t)blk * NRES * 3,
                                    out_p + (size_t)blk * NRES * 12);
  }
  k_copy<<<768, 256, 0, stream>>>(s, out_s, NRES * DS);
  #undef PF
}
